// UniGSA_layer_12403865551356
// MI455X (gfx1250) — matrix operand order repaired and checked by reading, not compiled, not run on hardware
//
#include <hip/hip_runtime.h>
#include <math.h>

typedef __attribute__((ext_vector_type(16))) __bf16 v16bf;
typedef __attribute__((ext_vector_type(8)))  float  v8f;
typedef __attribute__((ext_vector_type(4)))  unsigned int u32x4;
typedef __attribute__((ext_vector_type(8)))  int  i32x8;
typedef __attribute__((ext_vector_type(4)))  int  i32x4;

#define NN 10000          // nodes (fixed by reference)
#define EE 4000           // hyper-edges (fixed by reference)
#define NH 4
#define NF 64
#define ED 64
#define SLOPE 0.2f
#define KCHUNK 512        // K per wave = 16 wmma steps of K=32
#define KSPLIT ((NN + KCHUNK - 1) / KCHUNK)   // 20

// ---------------------------------------------------------------------------
// Kernel 1: p_pad[n,0..15] = edge_feat[n,:]·attn_m[h,:] for h<4 else 0 (bf16)
//           el[n,h]        = feat[n,h*64:]·attn_l[h,:]                 (f32)
// Zero-padding p to 16 columns makes the WMMA B-tile load unconditional.
// ---------------------------------------------------------------------------
__global__ __launch_bounds__(256) void proj_kernel(
    const float* __restrict__ feat, const float* __restrict__ edge_feat,
    const float* __restrict__ attn_l, const float* __restrict__ attn_m,
    __bf16* __restrict__ p_pad, float* __restrict__ el) {
  int idx = blockIdx.x * blockDim.x + threadIdx.x;   // n*16 + h
  if (idx >= NN * 16) return;
  int n = idx >> 4;
  int h = idx & 15;
  if (h < NH) {
    const float* ef = edge_feat + (size_t)n * ED;
    const float* am = attn_m + h * ED;
    const float* fs = feat + (size_t)n * (NH * NF) + h * NF;
    const float* al = attn_l + h * NF;
    float accm = 0.f, accl = 0.f;
#pragma unroll 8
    for (int d = 0; d < 64; ++d) {
      accm += ef[d] * am[d];
      accl += fs[d] * al[d];
    }
    p_pad[idx] = (__bf16)accm;
    el[(size_t)n * NH + h] = accl;
  } else {
    p_pad[idx] = (__bf16)0.0f;
  }
}

// ---------------------------------------------------------------------------
// Kernel 2: em(E,4) = H^T(E,N) @ p(N,16-padded)  via v_wmma_f32_16x16x32_bf16
// One wave per (e-tile, k-split). Fast path has NO per-element guards:
// 16 unconditional A loads (immediate offsets) + 16 unconditional B loads
// per WMMA. Clamped-address tail only in the last k-split.
// ---------------------------------------------------------------------------
__global__ __launch_bounds__(32) void em_wmma_kernel(
    const float* __restrict__ H, const __bf16* __restrict__ p_pad,
    float* __restrict__ em) {
  const int e0    = blockIdx.x * 16;
  const int lane  = threadIdx.x;
  const int mrow  = lane & 15;          // A row / B col / D col
  const int khalf = lane >> 4;

  const int k0   = blockIdx.y * KCHUNK;
  const int kend = (k0 + KCHUNK < NN) ? (k0 + KCHUNK) : NN;
  const int nfull = (kend - k0) >> 5;
  const int tail  = (kend - k0) & 31;

  const float*  hptr = H + (size_t)k0 * EE + (e0 + mrow);
  const __bf16* pptr = p_pad + (size_t)k0 * 16 + mrow;

  v8f acc = {};

  for (int it = 0; it < nfull; ++it) {
    v16bf a, b;
    // A (16x32 bf16): lanes 0-15 hold K={0..7,16..23}, lanes 16-31 {8..15,24..31}
#pragma unroll
    for (int i = 0; i < 16; ++i) {
      const int kk = ((i >> 3) << 4) | (khalf << 3) | (i & 7);
      a[i] = (__bf16)hptr[(size_t)kk * EE];
    }
    // B (32x16 bf16): lane = N col; lanes 0-15 K={0..7,16..23}, lanes 16-31 {8..15,24..31}
#pragma unroll
    for (int i = 0; i < 16; ++i) {
      const int kkb = ((i >> 3) << 4) | (khalf << 3) | (i & 7);
      b[i] = pptr[(size_t)kkb * 16];
    }
    acc = __builtin_amdgcn_wmma_f32_16x16x32_bf16(
        false, a, false, b, (short)0, acc, false, false);
    hptr += (size_t)32 * EE;
    pptr += 32 * 16;
  }

  if (tail) {   // only the last k-split; clamped addresses, branchless selects
    v16bf a, b;
#pragma unroll
    for (int i = 0; i < 16; ++i) {
      const int kk = ((i >> 3) << 4) | (khalf << 3) | (i & 7);
      const bool in = kk < tail;
      float hv = hptr[(size_t)(in ? kk : 0) * EE];
      a[i] = (__bf16)(in ? hv : 0.0f);
    }
#pragma unroll
    for (int i = 0; i < 16; ++i) {
      const int kkb = ((i >> 3) << 4) | (khalf << 3) | (i & 7);
      const bool in = kkb < tail;
      __bf16 pv = pptr[(size_t)(in ? kkb : 0) * 16];
      b[i] = in ? pv : (__bf16)0.0f;
    }
    acc = __builtin_amdgcn_wmma_f32_16x16x32_bf16(
        false, a, false, b, (short)0, acc, false, false);
  }

  // D layout: lane l holds D[M = j + 8*(l>>4), N = l&15] in acc[j]
  if (mrow < NH) {
#pragma unroll
    for (int j = 0; j < 8; ++j) {
      int e = e0 + j + 8 * khalf;
      atomicAdd(&em[(size_t)e * NH + mrow], acc[j]);
    }
  }
}

// ---------------------------------------------------------------------------
// Kernel 3: per node n — masked online softmax over its H row, then
//           out[n,:] = feat[n,:] * coef[n,h].
// em (62.5 KB) staged in LDS by the Tensor Data Mover (one DMA, wave 0 only),
// H row streamed as float4. 8 waves / block, one node per wave.
// ---------------------------------------------------------------------------
__global__ __launch_bounds__(256) void softmax_out_kernel(
    const float* __restrict__ H, const float* __restrict__ feat,
    const float* __restrict__ el, const float* __restrict__ em,
    float* __restrict__ out) {
  extern __shared__ float4 em_s[];   // EE entries of (em[e,0..3])

  // --- TDM: DMA em (8000 x 8B, one row) from global into LDS --------------
  int waveid = __builtin_amdgcn_readfirstlane((int)threadIdx.x) >> 5;
  if (waveid == 0) {
    unsigned long long ga = (unsigned long long)(uintptr_t)em;
    unsigned int lds_off = (unsigned int)(uintptr_t)(void*)em_s;  // LDS byte offset
    u32x4 g0;
    g0[0] = 1u;                                        // count=1 (valid user D#)
    g0[1] = lds_off;                                   // lds_addr
    g0[2] = (unsigned int)(ga & 0xffffffffu);          // global_addr[31:0]
    g0[3] = (unsigned int)((ga >> 32) & 0x01ffffffu)   // global_addr[56:32]
            | (2u << 30);                              // type = 2 (image)
    i32x8 g1;
    g1[0] = (int)(3u << 16);        // workgroup_mask=0, data_size=3 (8B)
    g1[1] = (int)(8000u << 16);     // tensor_dim0[15:0]=8000 (8B units)
    g1[2] = (int)(1u << 16);        // tensor_dim0[31:16]=0, tensor_dim1=1
    g1[3] = (int)(8000u << 16);     // tensor_dim1 hi=0, tile_dim0=8000
    g1[4] = 1;                      // tile_dim1=1, tile_dim2=0
    g1[5] = 8000;                   // tensor_dim0_stride low
    g1[6] = 0;
    g1[7] = 0;
    i32x4 gz4 = {0, 0, 0, 0};
    i32x8 gz8 = {0, 0, 0, 0, 0, 0, 0, 0};
    __builtin_amdgcn_tensor_load_to_lds(g0, g1, gz4, gz4, gz8, 0);
    __builtin_amdgcn_s_wait_tensorcnt(0);
  }
  __syncthreads();

  const int lane = threadIdx.x & 31;
  const int n = blockIdx.x * 8 + (threadIdx.x >> 5);
  if (n >= NN) return;

  const float* fr = feat + (size_t)n * (NH * NF);
  __builtin_prefetch(fr + lane * 8, 0, 3);   // warm the epilogue row

  const float4 elv = ((const float4*)el)[n];
  const float elh[4] = {elv.x, elv.y, elv.z, elv.w};

  float m[4], s[4];
#pragma unroll
  for (int h = 0; h < 4; ++h) { m[h] = -__builtin_inff(); s[h] = 0.f; }

  const float4* Hrow4 = (const float4*)(H + (size_t)n * EE);
  for (int i4 = lane; i4 < EE / 4; i4 += 32) {
    float4 h4 = Hrow4[i4];
    float hv[4] = {h4.x, h4.y, h4.z, h4.w};
#pragma unroll
    for (int c = 0; c < 4; ++c) {
      if (hv[c] != 0.f) {
        float4 ev4 = em_s[i4 * 4 + c];
        float ev[4] = {ev4.x, ev4.y, ev4.z, ev4.w};
#pragma unroll
        for (int h = 0; h < 4; ++h) {
          float x = elh[h] + ev[h];
          float v = (x > 0.f) ? x : SLOPE * x;        // leaky_relu
          float nm = fmaxf(m[h], v);
          float w  = (m[h] == nm) ? 1.0f : __expf(m[h] - nm);
          s[h] = s[h] * w + __expf(v - nm);
          m[h] = nm;
        }
      }
    }
  }

  // NaN-guarded butterfly reduction of (max, sumexp) across the wave
#pragma unroll
  for (int off = 16; off > 0; off >>= 1) {
#pragma unroll
    for (int h = 0; h < 4; ++h) {
      float om = __shfl_xor(m[h], off, 32);
      float os = __shfl_xor(s[h], off, 32);
      float nm = fmaxf(m[h], om);
      float w1 = (m[h] == nm) ? 1.0f : __expf(m[h] - nm);
      float w2 = (om   == nm) ? 1.0f : __expf(om   - nm);
      s[h] = s[h] * w1 + os * w2;
      m[h] = nm;
    }
  }

  // coef[h] = (sum over unmasked entries of the softmax) = Z * (1/Z)
  float coef[4];
#pragma unroll
  for (int h = 0; h < 4; ++h) coef[h] = s[h] * (1.0f / s[h]);

  float* orow = out + (size_t)n * (NH * NF);
#pragma unroll
  for (int h = 0; h < 4; ++h)
#pragma unroll
    for (int i = 0; i < 64; i += 32)
      orow[h * 64 + i + lane] = fr[h * 64 + i + lane] * coef[h];
}

// ---------------------------------------------------------------------------
extern "C" void kernel_launch(void* const* d_in, const int* in_sizes, int n_in,
                              void* d_out, int out_size, void* d_ws, size_t ws_size,
                              hipStream_t stream) {
  const float* feat      = (const float*)d_in[0];
  const float* edge_feat = (const float*)d_in[1];
  const float* H         = (const float*)d_in[2];
  const float* attn_l    = (const float*)d_in[3];
  const float* attn_m    = (const float*)d_in[4];
  float* out = (float*)d_out;

  // workspace layout (256B aligned slabs)
  char* ws = (char*)d_ws;
  size_t p_bytes  = ((size_t)NN * 16 * sizeof(__bf16) + 255) & ~(size_t)255;  // 320 KB
  size_t el_bytes = ((size_t)NN * NH * sizeof(float)  + 255) & ~(size_t)255;  // 160 KB
  __bf16* p_pad = (__bf16*)ws;
  float*  el    = (float*)(ws + p_bytes);
  float*  em    = (float*)(ws + p_bytes + el_bytes);                          // 64 KB

  (void)hipMemsetAsync(em, 0, (size_t)EE * NH * sizeof(float), stream);

  proj_kernel<<<(NN * 16 + 255) / 256, 256, 0, stream>>>(
      feat, edge_feat, attn_l, attn_m, p_pad, el);

  dim3 g(EE / 16, KSPLIT);                        // 250 x 20 waves
  em_wmma_kernel<<<g, 32, 0, stream>>>(H, p_pad, em);

  size_t lds = (size_t)EE * sizeof(float4);       // 62.5 KB
  softmax_out_kernel<<<NN / 8, 256, lds, stream>>>(H, feat, el, em, out);
}
